// AudioContextAggregation_55757265437247
// MI455X (gfx1250) — matrix operand order repaired and checked by reading, not compiled, not run on hardware
//
#include <hip/hip_runtime.h>
#include <hip/hip_bf16.h>

// ---------------------------------------------------------------------------
// AudioContextAggregation on gfx1250 (MI455X): windowed-gather cross-attention.
// All GEMMs (Q/K/V proj, Q*K^T, P*V, out proj) run on v_wmma_f32_16x16x32_bf16.
// GEMM A-tiles are double-buffer staged into LDS via the gfx1250 async-to-LDS
// path (GLOBAL_LOAD_ASYNC_TO_LDS_B128 / ASYNCcnt) when the builtin is exposed.
// LN / softmax / bias / residual stay fp32.
// ---------------------------------------------------------------------------

#if __has_builtin(__builtin_amdgcn_global_load_async_to_lds_b128)
#define ASYNC_LDS 1
#else
#define ASYNC_LDS 0
#endif

typedef __bf16 bf16_t;
typedef __attribute__((ext_vector_type(16))) __bf16 bf16x16;
typedef __attribute__((ext_vector_type(8)))  float  floatx8;
typedef int v4i __attribute__((vector_size(16)));
typedef __attribute__((address_space(1))) v4i* v4i_gptr;
typedef __attribute__((address_space(3))) v4i* v4i_lptr;

union FragU {
    bf16x16 v;
    uint4   q[2];
};

__device__ __forceinline__ unsigned short f2bf(float f) {
    unsigned int u = __float_as_uint(f);
    u += 0x7FFFu + ((u >> 16) & 1u);   // round-to-nearest-even
    return (unsigned short)(u >> 16);
}

// A-fragment (16x32 bf16, M x K): lanes 0-15 hold row M=lane with K chunks
// [k0, k0+8) and [k0+16, k0+24); lanes 16-31 hold same rows, chunks +8.
__device__ __forceinline__ bf16x16 load_frag_a(const unsigned short* A, int ld,
                                               int row0, int k0, int lane) {
    int r  = row0 + (lane & 15);
    int kc = k0 + ((lane >> 4) << 3);
    const unsigned short* p = A + (size_t)r * ld + kc;
    FragU f;
    f.q[0] = *(const uint4*)(p);
    f.q[1] = *(const uint4*)(p + 16);
    return f.v;
}

// Same A-fragment layout, reading a [rows x 32] LDS tile.
__device__ __forceinline__ bf16x16 load_frag_a_lds32(const unsigned short* base,
                                                     int row0, int lane) {
    int r  = row0 + (lane & 15);
    int kc = (lane >> 4) << 3;
    const unsigned short* p = base + r * 32 + kc;
    FragU f;
    f.q[0] = *(const uint4*)(p);
    f.q[1] = *(const uint4*)(p + 16);
    return f.v;
}

// B-fragment (32x16 bf16, K x N) for NT gemm where B[k][n] = W[n][k]:
// lanes 0-15 (col N=lane) hold W[N] K chunks [k0, k0+8) and [k0+16, k0+24);
// lanes 16-31 hold same cols, chunks +8.
__device__ __forceinline__ bf16x16 load_frag_b(const unsigned short* W, int ld,
                                               int n0, int k0, int lane) {
    int n  = n0 + (lane & 15);
    int kc = k0 + ((lane >> 4) << 3);
    const unsigned short* p = W + (size_t)n * ld + kc;
    FragU f;
    f.q[0] = *(const uint4*)(p);
    f.q[1] = *(const uint4*)(p + 16);
    return f.v;
}

__device__ __forceinline__ floatx8 wmma_bf16(bf16x16 a, bf16x16 b, floatx8 c) {
    return __builtin_amdgcn_wmma_f32_16x16x32_bf16(false, a, false, b,
                                                   (short)0, c, false, false);
}

__device__ __forceinline__ void wait_asynccnt0() {
#if ASYNC_LDS
#if __has_builtin(__builtin_amdgcn_s_wait_asynccnt)
    __builtin_amdgcn_s_wait_asynccnt(0);
#else
    asm volatile("s_wait_asynccnt 0x0" ::: "memory");
#endif
#endif
}

__device__ __forceinline__ float block_reduce_sum_256(float v, float* red) {
    int t = threadIdx.x;
    red[t] = v;
    __syncthreads();
    for (int s = 128; s > 0; s >>= 1) {
        if (t < s) red[t] += red[t + s];
        __syncthreads();
    }
    float r = red[0];
    __syncthreads();
    return r;
}

// ---------------- LayerNorm over hidden: (16384 rows x 1024) -> bf16 ---------
__global__ void ln_rows_kernel(const float* __restrict__ x,
                               const float* __restrict__ gamma,
                               const float* __restrict__ beta,
                               unsigned short* __restrict__ y) {
    __shared__ float red[256];
    int row = blockIdx.x;
    const float* xr = x + (size_t)row * 1024;
    float v[4];
    float s = 0.f;
#pragma unroll
    for (int i = 0; i < 4; i++) { v[i] = xr[threadIdx.x * 4 + i]; s += v[i]; }
    float mu = block_reduce_sum_256(s, red) * (1.f / 1024.f);
    float s2 = 0.f;
#pragma unroll
    for (int i = 0; i < 4; i++) { float d = v[i] - mu; s2 += d * d; }
    float var = block_reduce_sum_256(s2, red) * (1.f / 1024.f);
    float rs = rsqrtf(var + 1e-5f);
#pragma unroll
    for (int i = 0; i < 4; i++) {
        int c = threadIdx.x * 4 + i;
        y[(size_t)row * 1024 + c] = f2bf((v[i] - mu) * rs * gamma[c] + beta[c]);
    }
}

// -------- Windowed gather (zero-fill OOB frames) + LayerNorm -> bf16 --------
// Output rows padded 50 -> 64 per batch (pad rows written as zero).
__global__ void kv_gather_ln_kernel(const float* __restrict__ af,
                                    const int* __restrict__ frame_idx,
                                    const float* __restrict__ gamma,
                                    const float* __restrict__ beta,
                                    unsigned short* __restrict__ y, int T) {
    __shared__ float red[256];
    int m = blockIdx.x;
    int b = m >> 6;
    int j = m & 63;
    unsigned short* yr = y + (size_t)m * 1024;
    if (j >= 50) {  // pad rows: zeros (masked out of softmax anyway)
#pragma unroll
        for (int i = 0; i < 4; i++) yr[threadIdx.x * 4 + i] = 0;
        return;
    }
    int widx = j / 5, li = j - widx * 5;
    int fi = frame_idx[b] + widx - 4;             // WIN_LO = -4
    bool valid = (fi >= 0) && (fi < T);
    int fic = fi < 0 ? 0 : (fi > T - 1 ? T - 1 : fi);
    const float* xr = af + (((size_t)b * T + fic) * 5 + li) * 1024;
    float v[4];
    float s = 0.f;
#pragma unroll
    for (int i = 0; i < 4; i++) {
        v[i] = valid ? xr[threadIdx.x * 4 + i] : 0.f;
        s += v[i];
    }
    float mu = block_reduce_sum_256(s, red) * (1.f / 1024.f);
    float s2 = 0.f;
#pragma unroll
    for (int i = 0; i < 4; i++) { float d = v[i] - mu; s2 += d * d; }
    float var = block_reduce_sum_256(s2, red) * (1.f / 1024.f);
    float rs = rsqrtf(var + 1e-5f);
#pragma unroll
    for (int i = 0; i < 4; i++) {
        int c = threadIdx.x * 4 + i;
        yr[c] = f2bf((v[i] - mu) * rs * gamma[c] + beta[c]);
    }
}

// ---------------- fp32 -> bf16 weight conversion ----------------------------
__global__ void cvt_bf16_kernel(const float* __restrict__ src,
                                unsigned short* __restrict__ dst, int n) {
    int i = blockIdx.x * blockDim.x + threadIdx.x;
    if (i < n) dst[i] = f2bf(src[i]);
}

// ---------------- Generic NT GEMM: C[M,1024] = A[M,1024] * W[1024,1024]^T ---
// MODE 0: store bf16 row-major          (Q, K projections)
// MODE 1: store bf16 head-transposed Vt[b][h][hd=128][kv=64]   (V projection)
// MODE 2: store fp32 = acc + bias + residual                    (out proj)
// Block: 256 thr = 8 waves as 2 (row-groups) x 4 (col-groups);
// wave tile 32x64 => 8 WMMAs per 32-wide k-step.
// A block-tile (64x32 bf16, 4KB) is double-buffer staged into LDS, using the
// gfx1250 async-to-LDS path when available.
template <int MODE>
__global__ void gemm_bf16_nt(const unsigned short* __restrict__ A,
                             const unsigned short* __restrict__ W,
                             const float* __restrict__ bias,
                             void* __restrict__ out,
                             const float* __restrict__ resid) {
    const int K = 1024;
    __shared__ __align__(16) unsigned short abuf[2][64 * 32];

    int lane = threadIdx.x & 31;
    int w  = threadIdx.x >> 5;
    int wr = w & 1, wc = w >> 1;
    int m0 = blockIdx.y * 64 + wr * 32;
    int n0 = blockIdx.x * 256 + wc * 64;
    int row0 = blockIdx.y * 64;

    // staging: 256 threads x 16B == 64 rows x 32 cols bf16
    int srow   = threadIdx.x >> 2;
    int schunk = (threadIdx.x & 3) * 8;
    const unsigned short* gsrc = A + (size_t)(row0 + srow) * K + schunk;
    unsigned short* lslot0 = &abuf[0][srow * 32 + schunk];
    unsigned short* lslot1 = &abuf[1][srow * 32 + schunk];

    auto stage = [&](int bufi, int k0) {
        unsigned short* dst = bufi ? lslot1 : lslot0;
#if ASYNC_LDS
        __builtin_amdgcn_global_load_async_to_lds_b128(
            (v4i_gptr)(gsrc + k0), (v4i_lptr)dst, 0, 0);
#else
        *(uint4*)dst = *(const uint4*)(gsrc + k0);
#endif
    };

    stage(0, 0);
    floatx8 acc[2][4] = {};
    for (int k0 = 0, it = 0; k0 < K; k0 += 32, ++it) {
        int cur = it & 1;
        wait_asynccnt0();
        __syncthreads();  // buf[cur] visible to all waves; prev compute done
        if (k0 + 32 < K) stage(cur ^ 1, k0 + 32);

        bf16x16 a0 = load_frag_a_lds32(abuf[cur], wr * 32, lane);
        bf16x16 a1 = load_frag_a_lds32(abuf[cur], wr * 32 + 16, lane);
#pragma unroll
        for (int nt = 0; nt < 4; nt++) {
            bf16x16 bfr = load_frag_b(W, K, n0 + nt * 16, k0, lane);
            acc[0][nt] = wmma_bf16(a0, bfr, acc[0][nt]);
            acc[1][nt] = wmma_bf16(a1, bfr, acc[1][nt]);
        }
    }

#pragma unroll
    for (int rt = 0; rt < 2; rt++) {
        int rbase = m0 + rt * 16 + ((lane >> 4) << 3);
#pragma unroll
        for (int nt = 0; nt < 4; nt++) {
            int col = n0 + nt * 16 + (lane & 15);
            float bc = bias ? bias[col] : 0.f;
#pragma unroll
            for (int r = 0; r < 8; r++) {
                float val = acc[rt][nt][r] + bc;
                int row = rbase + r;
                if (MODE == 0) {
                    ((unsigned short*)out)[(size_t)row * 1024 + col] = f2bf(val);
                } else if (MODE == 1) {
                    int b = row >> 6, kvn = row & 63;
                    int h = col >> 7, d = col & 127;
                    ((unsigned short*)out)[(((size_t)(b * 8 + h) * 128 + d) << 6) + kvn] =
                        f2bf(val);
                } else {
                    size_t idx = (size_t)row * 1024 + col;
                    ((float*)out)[idx] = val + resid[idx];
                }
            }
        }
    }
}

// ---------------- Fused attention per (b, h, 16-q-row tile) -----------------
// scores(16x64) = Q(16x128) * K^T -> masked softmax(50) -> ctx = P * V(64x128)
__global__ void attn_kernel(const unsigned short* __restrict__ Q,
                            const unsigned short* __restrict__ Kb,
                            const unsigned short* __restrict__ Vt,
                            unsigned short* __restrict__ ctx) {
    __shared__ __align__(16) unsigned short plds[4][16 * 64];
    int lane = threadIdx.x & 31;
    int w  = threadIdx.x >> 5;
    int gw = blockIdx.x * 4 + w;     // global wave id, 8192 total
    int b  = gw >> 9;                // / (H*64)
    int h  = (gw >> 6) & 7;
    int m0 = (gw & 63) << 4;

    const unsigned short* Qh = Q  + (size_t)b * 1024 * 1024 + h * 128;
    const unsigned short* Kh = Kb + (size_t)b * 64 * 1024 + h * 128;
    const unsigned short* Vh = Vt + (size_t)(b * 8 + h) * 128 * 64;

    // --- Q * K^T over hd=128 (4 k-steps), 4 kv col-tiles ---
    bf16x16 qf[4];
#pragma unroll
    for (int kt = 0; kt < 4; kt++) qf[kt] = load_frag_a(Qh, 1024, m0, kt * 32, lane);

    floatx8 s[4] = {};
#pragma unroll
    for (int kt = 0; kt < 4; kt++) {
#pragma unroll
        for (int nt = 0; nt < 4; nt++) {
            bf16x16 kb = load_frag_b(Kh, 1024, nt * 16, kt * 32, lane);
            s[nt] = wmma_bf16(qf[kt], kb, s[nt]);
        }
    }

    // --- masked softmax over 50 kv positions, fp32 ---
    const float scale = 0.088388347648318447f;  // 1/sqrt(128)
    int ncol = lane & 15;
    float pr[4][8];
#pragma unroll
    for (int r = 0; r < 8; r++) {
        float mx = -1e30f;
#pragma unroll
        for (int t = 0; t < 4; t++) {
            float v = (t * 16 + ncol < 50) ? s[t][r] : -1e30f;
            pr[t][r] = v;
            mx = fmaxf(mx, v);
        }
        mx = fmaxf(mx, __shfl_xor(mx, 1, 32));
        mx = fmaxf(mx, __shfl_xor(mx, 2, 32));
        mx = fmaxf(mx, __shfl_xor(mx, 4, 32));
        mx = fmaxf(mx, __shfl_xor(mx, 8, 32));
        float sum = 0.f;
#pragma unroll
        for (int t = 0; t < 4; t++) {
            float e = (pr[t][r] > -1e29f) ? __expf((pr[t][r] - mx) * scale) : 0.f;
            pr[t][r] = e;
            sum += e;
        }
        sum += __shfl_xor(sum, 1, 32);
        sum += __shfl_xor(sum, 2, 32);
        sum += __shfl_xor(sum, 4, 32);
        sum += __shfl_xor(sum, 8, 32);
        float inv = 1.f / sum;
#pragma unroll
        for (int t = 0; t < 4; t++) pr[t][r] *= inv;
    }

    // --- relayout P (C-frag) -> A-frag via LDS ---
    unsigned short* pw = plds[w];
    int rowoff = (lane >> 4) << 3;
#pragma unroll
    for (int r = 0; r < 8; r++)
#pragma unroll
        for (int t = 0; t < 4; t++)
            pw[(rowoff + r) * 64 + t * 16 + ncol] = f2bf(pr[t][r]);
    asm volatile("s_wait_dscnt 0x0" ::: "memory");

    bf16x16 pf[2];
#pragma unroll
    for (int ka = 0; ka < 2; ka++) {
        int rr = lane & 15;
        int kc = ka * 32 + ((lane >> 4) << 3);
        FragU f;
        f.q[0] = *(const uint4*)(pw + rr * 64 + kc);
        f.q[1] = *(const uint4*)(pw + rr * 64 + kc + 16);
        pf[ka] = f.v;
    }

    // --- ctx(16x128) = P(16x64) * V(64x128); Vt is [hd][kv] so B loads are contiguous
    floatx8 c[8] = {};
#pragma unroll
    for (int ka = 0; ka < 2; ka++) {
#pragma unroll
        for (int nt = 0; nt < 8; nt++) {
            bf16x16 vb = load_frag_b(Vh, 64, nt * 16, ka * 32, lane);
            c[nt] = wmma_bf16(pf[ka], vb, c[nt]);
        }
    }

#pragma unroll
    for (int nt = 0; nt < 8; nt++) {
        int col = h * 128 + nt * 16 + ncol;
#pragma unroll
        for (int r = 0; r < 8; r++) {
            int row = b * 1024 + m0 + rowoff + r;
            ctx[(size_t)row * 1024 + col] = f2bf(c[nt][r]);
        }
    }
}

// ---------------------------------------------------------------------------
extern "C" void kernel_launch(void* const* d_in, const int* in_sizes, int n_in,
                              void* d_out, int out_size, void* d_ws, size_t ws_size,
                              hipStream_t stream) {
    const float* hidden = (const float*)d_in[0];
    const float* af     = (const float*)d_in[1];
    const int*   fidx   = (const int*)d_in[2];
    const float* q_g    = (const float*)d_in[3];
    const float* q_b    = (const float*)d_in[4];
    const float* kv_g   = (const float*)d_in[5];
    const float* kv_b   = (const float*)d_in[6];
    const float* in_w   = (const float*)d_in[7];
    const float* in_b   = (const float*)d_in[8];
    const float* out_w  = (const float*)d_in[9];
    const float* out_b  = (const float*)d_in[10];
    float* out = (float*)d_out;

    const int B = 16, D = 1024, Lq = 1024, H = 8;
    const int T = in_sizes[1] / (B * 5 * D);   // 200

    char* ws = (char*)d_ws;
    size_t off = 0;
    auto wsalloc = [&](size_t bytes) -> void* {
        void* p = ws + off;
        off += (bytes + 255) & ~(size_t)255;
        return p;
    };
    unsigned short* qln  = (unsigned short*)wsalloc((size_t)B * Lq * D * 2);
    unsigned short* kvln = (unsigned short*)wsalloc((size_t)B * 64 * D * 2);
    unsigned short* wbf  = (unsigned short*)wsalloc((size_t)4 * D * D * 2);
    unsigned short* Qb   = (unsigned short*)wsalloc((size_t)B * Lq * D * 2);
    unsigned short* Kbuf = (unsigned short*)wsalloc((size_t)B * 64 * D * 2);
    unsigned short* Vt   = (unsigned short*)wsalloc((size_t)B * H * 128 * 64 * 2);
    unsigned short* ctx  = (unsigned short*)wsalloc((size_t)B * Lq * D * 2);

    ln_rows_kernel<<<B * Lq, 256, 0, stream>>>(hidden, q_g, q_b, qln);
    kv_gather_ln_kernel<<<B * 64, 256, 0, stream>>>(af, fidx, kv_g, kv_b, kvln, T);
    cvt_bf16_kernel<<<(3 * D * D) / 256, 256, 0, stream>>>(in_w, wbf, 3 * D * D);
    cvt_bf16_kernel<<<(D * D) / 256, 256, 0, stream>>>(out_w, wbf + (size_t)3 * D * D, D * D);

    dim3 gq(4, (B * Lq) / 64);     // N/256 x M/64
    dim3 gkv(4, (B * 64) / 64);
    gemm_bf16_nt<0><<<gq, 256, 0, stream>>>(qln, wbf, in_b, Qb, nullptr);
    gemm_bf16_nt<0><<<gkv, 256, 0, stream>>>(kvln, wbf + (size_t)D * D, in_b + D, Kbuf, nullptr);
    gemm_bf16_nt<1><<<gkv, 256, 0, stream>>>(kvln, wbf + (size_t)2 * D * D, in_b + 2 * D, Vt, nullptr);

    attn_kernel<<<(B * H * 64) / 4, 128, 0, stream>>>(Qb, Kbuf, Vt, ctx);

    gemm_bf16_nt<2><<<gq, 256, 0, stream>>>(ctx, wbf + (size_t)3 * D * D, out_b, out, hidden);
}
